// FeatureProp_43447889167194
// MI455X (gfx1250) — compile-verified, matrix operand order checked by reading, not run on hardware
//
#include <hip/hip_runtime.h>
#include <hip/hip_bf16.h>

typedef __attribute__((ext_vector_type(16))) _Float16 v16h;
typedef __attribute__((ext_vector_type(8)))  float    v8f;

#define B_   16
#define N1_  4096
#define N2_  1024
#define CIN  256
#define CSK  64
#define CCAT 320
#define COUT 256
#define TM   16
#define NROW (B_ * N1_)   // 65536 flattened points

// ---------------------------------------------------------------------------
// Kernel 0: stage W (f32 -> f16) into workspace, zero the stats slots.
// ---------------------------------------------------------------------------
__global__ __launch_bounds__(256) void fp_prep(const float* __restrict__ W,
                                               _Float16* __restrict__ Wh,
                                               float* __restrict__ stats) {
  const int i = blockIdx.x * 256 + threadIdx.x;
  if (i < 2 * COUT) stats[i] = 0.f;
  if (i < COUT * CCAT) Wh[i] = (_Float16)W[i];
}

// ---------------------------------------------------------------------------
// Kernel 1: per 16-point tile: 3-NN search + interpolation + WMMA GEMM.
// Block = 256 threads = 8 waves.  Grid = (B*N1)/16 = 4096 blocks.
// ---------------------------------------------------------------------------
__global__ __launch_bounds__(256) void fp_main(const float* __restrict__ xyz1,
                                               const float* __restrict__ xyz2,
                                               const float* __restrict__ feat1,
                                               const float* __restrict__ feat2,
                                               const _Float16* __restrict__ Wh,
                                               const float* __restrict__ bias,
                                               float* __restrict__ y) {
  __shared__ float    s_x2[N2_ * 3];      // 12 KB: xyz2 for this batch
  __shared__ float    s_x1[TM * 3];
  __shared__ float    s_bd[256 * 3];      // per-thread top-3 distances
  __shared__ int      s_bi[256 * 3];      // per-thread top-3 indices
  __shared__ int      s_idx[TM * 3];      // final 3-NN indices per point
  __shared__ _Float16 s_cat[TM * CCAT];   // 10 KB: [feat1 | interp] tile, f16

  const int tid   = threadIdx.x;
  const int blk   = blockIdx.x;           // 0..4095
  const int b     = blk >> 8;             // / (N1/TM = 256)
  const int tile  = blk & 255;
  const int r0    = tile * TM;
  const long grow0 = (long)b * N1_ + r0;  // flat row base in [B*N1]

  // ---- stage xyz into LDS ----
  for (int i = tid; i < N2_ * 3; i += 256)
    s_x2[i] = xyz2[(long)b * N2_ * 3 + i];
  if (tid < TM * 3) s_x1[tid] = xyz1[grow0 * 3 + tid];
  __syncthreads();

  // ---- top-3 nearest neighbors: 16 threads per point, strided scan ----
  {
    const int p = tid >> 4, sub = tid & 15;
    const float px = s_x1[p * 3 + 0], py = s_x1[p * 3 + 1], pz = s_x1[p * 3 + 2];
    float d0 = 3.4e38f, d1 = 3.4e38f, d2 = 3.4e38f;
    int   i0 = 0, i1 = 0, i2 = 0;
    for (int j = sub; j < N2_; j += 16) {
      const float dx = s_x2[j * 3 + 0] - px;
      const float dy = s_x2[j * 3 + 1] - py;
      const float dz = s_x2[j * 3 + 2] - pz;
      const float d  = dx * dx + dy * dy + dz * dz;
      if (d < d0)      { d2 = d1; i2 = i1; d1 = d0; i1 = i0; d0 = d; i0 = j; }
      else if (d < d1) { d2 = d1; i2 = i1; d1 = d;  i1 = j; }
      else if (d < d2) { d2 = d;  i2 = j; }
    }
    s_bd[tid * 3 + 0] = d0; s_bd[tid * 3 + 1] = d1; s_bd[tid * 3 + 2] = d2;
    s_bi[tid * 3 + 0] = i0; s_bi[tid * 3 + 1] = i1; s_bi[tid * 3 + 2] = i2;
  }
  __syncthreads();
  if (tid < TM) {  // merge 48 candidates -> final top-3 (one thread per point)
    float d0 = 3.4e38f, d1 = 3.4e38f, d2 = 3.4e38f;
    int   i0 = 0, i1 = 0, i2 = 0;
    const int base = tid * 16 * 3;
    for (int q = 0; q < 48; ++q) {
      const float d = s_bd[base + q];
      const int   i = s_bi[base + q];
      if (d < d0)      { d2 = d1; i2 = i1; d1 = d0; i1 = i0; d0 = d; i0 = i; }
      else if (d < d1) { d2 = d1; i2 = i1; d1 = d;  i1 = i; }
      else if (d < d2) { d2 = d;  i2 = i; }
    }
    s_idx[tid * 3 + 0] = i0; s_idx[tid * 3 + 1] = i1; s_idx[tid * 3 + 2] = i2;
  }
  __syncthreads();

  // ---- build concat tile [feat1 | mean(3 gathered feat2 rows)] in f16 ----
  {
    const float* f2b = feat2 + (long)b * N2_ * CIN;
    for (int i = tid; i < TM * CIN; i += 256) {
      const int p = i >> 8, c = i & 255;
      const float v = (f2b[(long)s_idx[p * 3 + 0] * CIN + c] +
                       f2b[(long)s_idx[p * 3 + 1] * CIN + c] +
                       f2b[(long)s_idx[p * 3 + 2] * CIN + c]) * (1.0f / 3.0f);
      s_cat[p * CCAT + CSK + c] = (_Float16)v;
    }
    for (int i = tid; i < TM * CSK; i += 256) {
      const int p = i >> 6, c = i & 63;
      s_cat[p * CCAT + c] = (_Float16)feat1[(grow0 + p) * CSK + c];
    }
  }
  __syncthreads();

  // ---- GEMM: [16 x 320] (LDS, f16) x [320 x 256] (Wh^T) via WMMA ----
  // v_wmma_f32_16x16x32_f16; 8 waves x 2 n-tiles x 10 K-steps.
  // 16-bit A 16x32 lane layout (ISA 7.12.2): lane half h holds
  // K = {h*8..h*8+7} in elements 0..7 and K = {16+h*8..} in elements 8..15.
  const int lane = tid & 31;
  const int wave = tid >> 5;
  const int rm   = lane & 15;   // A row / B column-in-tile / D column-in-tile
  const int h    = lane >> 4;   // K-half for A/B; M-half for D

  for (int t = 0; t < 2; ++t) {
    const int col = (wave * 2 + t) * 16 + rm;  // output channel
    v8f acc = {};
    for (int ks = 0; ks < CCAT / 32; ++ks) {
      const int k0 = ks * 32;
      v16h a, w;
#pragma unroll
      for (int e = 0; e < 16; ++e) {
        const int kl = k0 + h * 8 + (e < 8 ? e : e + 8);
        a[e] = s_cat[rm * CCAT + kl];          // A[row=rm][k]
        w[e] = Wh[(long)col * CCAT + kl];      // B[k][n] = W[col][k]
      }
      acc = __builtin_amdgcn_wmma_f32_16x16x32_f16(
          /*neg_a=*/false, a, /*neg_b=*/false, w,
          /*c_mod=*/(short)0, acc, /*reuse_a=*/false, /*reuse_b=*/false);
    }
    const float bv = bias[col];
#pragma unroll
    for (int r = 0; r < 8; ++r) {
      const int m = r + h * 8;                 // D row within tile
      y[(grow0 + m) * COUT + col] = acc[r] + bv;
    }
  }
}

// ---------------------------------------------------------------------------
// Kernel 2a: per-channel partial sum / sumsq over 256-row slabs (coalesced,
// deterministic: each thread owns one channel, sequential accumulation).
// Grid = 256 blocks (row slabs), 256 threads (channels).
// ---------------------------------------------------------------------------
__global__ __launch_bounds__(256) void fp_stats1(const float* __restrict__ y,
                                                 float* __restrict__ psum,
                                                 float* __restrict__ psum2) {
  const int c   = threadIdx.x;
  const int blk = blockIdx.x;
  float s = 0.f, s2 = 0.f;
  const long rbase = (long)blk * (NROW / 256);
  for (int r = 0; r < NROW / 256; ++r) {
    const float v = y[(rbase + r) * COUT + c];
    s += v; s2 += v * v;
  }
  psum [blk * COUT + c] = s;
  psum2[blk * COUT + c] = s2;
}

// Kernel 2b: combine 256 partials per channel (fixed order => deterministic).
__global__ __launch_bounds__(256) void fp_stats2(const float* __restrict__ psum,
                                                 const float* __restrict__ psum2,
                                                 float* __restrict__ stats) {
  const int c = threadIdx.x;
  float s = 0.f, s2 = 0.f;
  for (int j = 0; j < 256; ++j) {
    s  += psum [j * COUT + c];
    s2 += psum2[j * COUT + c];
  }
  stats[c] = s;
  stats[COUT + c] = s2;
}

// ---------------------------------------------------------------------------
// Kernel 3: BatchNorm (batch stats) + ReLU, in place on d_out.
// ---------------------------------------------------------------------------
__global__ __launch_bounds__(256) void fp_bn_relu(float* __restrict__ y,
                                                  const float* __restrict__ stats,
                                                  const float* __restrict__ gamma,
                                                  const float* __restrict__ beta) {
  const long i = (long)blockIdx.x * 256 + threadIdx.x;
  const int  c = (int)(i & (COUT - 1));
  const float inv_n = 1.0f / (float)NROW;
  const float mean = stats[c] * inv_n;
  const float var  = stats[COUT + c] * inv_n - mean * mean;
  const float v = (y[i] - mean) * rsqrtf(var + 1e-5f) * gamma[c] + beta[c];
  y[i] = fmaxf(v, 0.f);
}

// ---------------------------------------------------------------------------
extern "C" void kernel_launch(void* const* d_in, const int* in_sizes, int n_in,
                              void* d_out, int out_size, void* d_ws, size_t ws_size,
                              hipStream_t stream) {
  (void)in_sizes; (void)n_in; (void)out_size; (void)ws_size;
  const float* xyz1  = (const float*)d_in[0];
  const float* xyz2  = (const float*)d_in[1];
  const float* feat1 = (const float*)d_in[2];
  const float* feat2 = (const float*)d_in[3];
  const float* W     = (const float*)d_in[4];
  const float* bias  = (const float*)d_in[5];
  const float* gamma = (const float*)d_in[6];
  const float* beta  = (const float*)d_in[7];
  float* y = (float*)d_out;

  // workspace layout: [stats 512 f32][psum 64K f32][psum2 64K f32][Wh 80K f16]
  float*    stats = (float*)d_ws;
  float*    psum  = stats + 2 * COUT;
  float*    psum2 = psum + 256 * COUT;
  _Float16* Wh    = (_Float16*)(psum2 + 256 * COUT);

  fp_prep<<<(COUT * CCAT + 255) / 256, 256, 0, stream>>>(W, Wh, stats);
  fp_main<<<NROW / TM, 256, 0, stream>>>(xyz1, xyz2, feat1, feat2, Wh, bias, y);
  fp_stats1<<<256, 256, 0, stream>>>(y, psum, psum2);
  fp_stats2<<<1, 256, 0, stream>>>(psum, psum2, stats);
  fp_bn_relu<<<(int)((long)NROW * COUT / 256), 256, 0, stream>>>(y, stats, gamma, beta);
}
